// molGraphRepresentation_12154757448185
// MI455X (gfx1250) — matrix operand order repaired and checked by reading, not compiled, not run on hardware
//
#include <hip/hip_runtime.h>
#include <hip/hip_bf16.h>

// ---------------------------------------------------------------------------
// Problem constants (from reference)
// ---------------------------------------------------------------------------
#define NODE_DIM 78
#define DIM      128
#define HEADS    4
#define LAYERS   3
#define HID      1024
#define SLOPE    0.2f
#define EPS_A    1e-16f

typedef _Float16 v16h __attribute__((ext_vector_type(16)));
typedef _Float16 v8h  __attribute__((ext_vector_type(8)));
typedef float    v8f  __attribute__((ext_vector_type(8)));

// ---------------------------------------------------------------------------
// f32 -> f16 conversion with K padding (row-major [M,K] -> [M,Kpad])
// ---------------------------------------------------------------------------
__global__ void conv_pad_kernel(const float* __restrict__ in, _Float16* __restrict__ out,
                                int M, int K, int Kpad) {
    long long i = (long long)blockIdx.x * blockDim.x + threadIdx.x;
    if (i >= (long long)M * Kpad) return;
    int m = (int)(i / Kpad), k = (int)(i % Kpad);
    out[i] = (k < K) ? (_Float16)in[(size_t)m * K + k] : (_Float16)0.0f;
}

// f32 [K,N] row-major -> f16 [N,Kpad] (transposed, K zero-padded)
__global__ void conv_trans_kernel(const float* __restrict__ in, _Float16* __restrict__ out,
                                  int K, int N, int Kpad) {
    long long i = (long long)blockIdx.x * blockDim.x + threadIdx.x;
    if (i >= (long long)N * Kpad) return;
    int n = (int)(i / Kpad), k = (int)(i % Kpad);
    out[i] = (k < K) ? (_Float16)in[(size_t)k * N + n] : (_Float16)0.0f;
}

// ---------------------------------------------------------------------------
// WMMA GEMM: C[M,N] = act(A[M,K] * B[K,N] + bias[N])
//   Ah: f16 row-major [M,Kpad]   (activations, pre-converted, zero-padded)
//   Bt: f16 col-major [N,Kpad]   (weights, pre-transposed)
// One wave computes a 16x64 strip: one A fragment reused across 4 WMMA tiles.
// M % 16 == 0, N % 64 == 0, Kpad % 32 == 0 (guaranteed by callers).
// ---------------------------------------------------------------------------
__device__ __forceinline__ v16h ld_frag(const _Float16* p0, const _Float16* p1) {
    v8h lo = *(const v8h*)p0;
    v8h hi = *(const v8h*)p1;
    return __builtin_shufflevector(lo, hi, 0,1,2,3,4,5,6,7,8,9,10,11,12,13,14,15);
}

__global__ __launch_bounds__(256)
void wmma_gemm16_kernel(const _Float16* __restrict__ Ah, const _Float16* __restrict__ Bt,
                        const float* __restrict__ bias, float* __restrict__ C,
                        int M, int N, int Kpad, int relu) {
    const int lane = threadIdx.x & 31;
    const int wid  = blockIdx.x * (blockDim.x >> 5) + (threadIdx.x >> 5);
    const int tilesN4 = N >> 6;                 // 64-column strips
    const int tilesM  = M >> 4;
    if (wid >= tilesM * tilesN4) return;
    const int tm  = wid / tilesN4;
    const int tn4 = wid % tilesN4;

    const int hi  = lane >> 4;                  // half-wave select
    const int l16 = lane & 15;

    const _Float16* arow = Ah + (size_t)(tm * 16 + l16) * Kpad;
    const _Float16* bc0  = Bt + (size_t)(tn4 * 64 +  0 + l16) * Kpad + hi * 16;
    const _Float16* bc1  = Bt + (size_t)(tn4 * 64 + 16 + l16) * Kpad + hi * 16;
    const _Float16* bc2  = Bt + (size_t)(tn4 * 64 + 32 + l16) * Kpad + hi * 16;
    const _Float16* bc3  = Bt + (size_t)(tn4 * 64 + 48 + l16) * Kpad + hi * 16;

    v8f acc0 = {}, acc1 = {}, acc2 = {}, acc3 = {};
    for (int kc = 0; kc < Kpad; kc += 32) {
        // A 16-bit fragment: element e -> K = e + hi*16
        v16h a  = ld_frag(arow + kc + hi * 16, arow + kc + hi * 16 + 8);
        // B 16-bit fragment: 16 contiguous K values starting at kc + hi*16
        v16h b0 = ld_frag(bc0 + kc, bc0 + kc + 8);
        v16h b1 = ld_frag(bc1 + kc, bc1 + kc + 8);
        v16h b2 = ld_frag(bc2 + kc, bc2 + kc + 8);
        v16h b3 = ld_frag(bc3 + kc, bc3 + kc + 8);
        acc0 = __builtin_amdgcn_wmma_f32_16x16x32_f16(false, a, false, b0, (short)0, acc0, false, false);
        acc1 = __builtin_amdgcn_wmma_f32_16x16x32_f16(false, a, false, b1, (short)0, acc1, false, false);
        acc2 = __builtin_amdgcn_wmma_f32_16x16x32_f16(false, a, false, b2, (short)0, acc2, false, false);
        acc3 = __builtin_amdgcn_wmma_f32_16x16x32_f16(false, a, false, b3, (short)0, acc3, false, false);
    }

    // Store: C layout VGPR v -> row tm*16 + hi*8 + v, col = strip base + l16
#pragma unroll
    for (int j = 0; j < 4; ++j) {
        v8f acc = (j == 0) ? acc0 : (j == 1) ? acc1 : (j == 2) ? acc2 : acc3;
        int cn = tn4 * 64 + j * 16 + l16;
        float bv = bias ? bias[cn] : 0.0f;
#pragma unroll
        for (int v = 0; v < 8; ++v) {
            int cm = tm * 16 + hi * 8 + v;
            float val = acc[v] + bv;
            if (relu) val = fmaxf(val, 0.0f);
            C[(size_t)cm * N + cn] = val;
        }
    }
}

// ---------------------------------------------------------------------------
// Fills
// ---------------------------------------------------------------------------
__global__ void fill_f32_kernel(float* p, float v, size_t n) {
    size_t i = (size_t)blockIdx.x * blockDim.x + threadIdx.x;
    if (i < n) p[i] = v;
}
__global__ void fill_i32_kernel(int* p, int v, size_t n) {
    size_t i = (size_t)blockIdx.x * blockDim.x + threadIdx.x;
    if (i < n) p[i] = v;
}

// ---------------------------------------------------------------------------
// Attention coefficients: one wave per (node, head)
// ---------------------------------------------------------------------------
__global__ __launch_bounds__(256)
void att_coef_kernel(const float* __restrict__ hp,
                     const float* __restrict__ att_s, const float* __restrict__ att_d,
                     float* __restrict__ a_s, float* __restrict__ a_d, int Nn) {
    const int lane = threadIdx.x & 31;
    const int wid  = blockIdx.x * (blockDim.x >> 5) + (threadIdx.x >> 5);
    if (wid >= Nn * HEADS) return;
    const int n = wid >> 2, hh = wid & 3;
    const float* row = hp + ((size_t)n * HEADS + hh) * DIM;
    float ss = 0.f, sd = 0.f;
#pragma unroll
    for (int c = lane; c < DIM; c += 32) {
        float v = row[c];
        ss += v * att_s[hh * DIM + c];
        sd += v * att_d[hh * DIM + c];
    }
#pragma unroll
    for (int o = 16; o > 0; o >>= 1) {
        ss += __shfl_xor(ss, o, 32);
        sd += __shfl_xor(sd, o, 32);
    }
    if (lane == 0) { a_s[wid] = ss; a_d[wid] = sd; }
}

// ---------------------------------------------------------------------------
// Edge kernels. Edges e < E use edge_index; e >= E are self loops.
// ---------------------------------------------------------------------------
__device__ __forceinline__ float edge_score(const float* a_s, const float* a_d,
                                            int s, int d, int hh) {
    float sc = a_s[s * HEADS + hh] + a_d[d * HEADS + hh];
    return sc > 0.0f ? sc : sc * SLOPE;     // leaky_relu
}

__device__ __forceinline__ void atomic_max_f32(int* addr, float val) {
    int iv = __float_as_int(val);
    if (iv >= 0) atomicMax(addr, iv);
    else         atomicMin((unsigned int*)addr, (unsigned int)iv);
}

__global__ void edge_max_kernel(const int* __restrict__ src, const int* __restrict__ dst,
                                int E, int Nn,
                                const float* __restrict__ a_s, const float* __restrict__ a_d,
                                int* __restrict__ m_bits) {
    long long tid = (long long)blockIdx.x * blockDim.x + threadIdx.x;
    if (tid >= (long long)(E + Nn) * HEADS) return;
    int e = (int)(tid >> 2), hh = (int)(tid & 3);
    int s, d;
    if (e < E) { s = src[e]; d = dst[e]; } else { s = d = e - E; }
    atomic_max_f32(m_bits + d * HEADS + hh, edge_score(a_s, a_d, s, d, hh));
}

__global__ void edge_denom_kernel(const int* __restrict__ src, const int* __restrict__ dst,
                                  int E, int Nn,
                                  const float* __restrict__ a_s, const float* __restrict__ a_d,
                                  const int* __restrict__ m_bits, float* __restrict__ denom) {
    long long tid = (long long)blockIdx.x * blockDim.x + threadIdx.x;
    if (tid >= (long long)(E + Nn) * HEADS) return;
    int e = (int)(tid >> 2), hh = (int)(tid & 3);
    int s, d;
    if (e < E) { s = src[e]; d = dst[e]; } else { s = d = e - E; }
    float sc = edge_score(a_s, a_d, s, d, hh);
    float m  = __int_as_float(m_bits[d * HEADS + hh]);
    atomicAdd(denom + d * HEADS + hh, __expf(sc - m));
}

// out[dst,h,:] += alpha * hp[src,h,:]; 128 threads per edge (4 heads x 32 lanes).
__global__ __launch_bounds__(128)
void edge_acc_kernel(const int* __restrict__ src, const int* __restrict__ dst,
                     int E, int Nn,
                     const float* __restrict__ a_s, const float* __restrict__ a_d,
                     const int* __restrict__ m_bits, const float* __restrict__ denom,
                     const float* __restrict__ hp, float* __restrict__ out) {
    int e = blockIdx.x;
    int hh = threadIdx.x >> 5, lane = threadIdx.x & 31;
    int s, d;
    if (e < E) { s = src[e]; d = dst[e]; } else { s = d = e - E; }
    float sc = edge_score(a_s, a_d, s, d, hh);
    float m  = __int_as_float(m_bits[d * HEADS + hh]);
    float alpha = __expf(sc - m) / (denom[d * HEADS + hh] + EPS_A);

    const float4 v = ((const float4*)(hp + ((size_t)s * HEADS + hh) * DIM))[lane];
    float* o = out + ((size_t)d * HEADS + hh) * DIM + lane * 4;
    atomicAdd(o + 0, v.x * alpha);
    atomicAdd(o + 1, v.y * alpha);
    atomicAdd(o + 2, v.z * alpha);
    atomicAdd(o + 3, v.w * alpha);
}

__global__ void head_mean_kernel(const float* __restrict__ out, const float* __restrict__ b,
                                 float* __restrict__ h, int Nn) {
    long long tid = (long long)blockIdx.x * blockDim.x + threadIdx.x;
    if (tid >= (long long)Nn * DIM) return;
    int n = (int)(tid >> 7), c = (int)(tid & 127);
    const float* base = out + (size_t)n * HEADS * DIM + c;
    float acc = base[0] + base[DIM] + base[2 * DIM] + base[3 * DIM];
    h[tid] = acc * (1.0f / HEADS) + b[c];
}

// ---------------------------------------------------------------------------
// Graph pooling
// ---------------------------------------------------------------------------
__global__ __launch_bounds__(128)
void pool_acc_kernel(const float* __restrict__ h, const int* __restrict__ batch,
                     float* __restrict__ pooled, float* __restrict__ counts, int Nn) {
    int n = blockIdx.x;
    if (n >= Nn) return;
    int g = batch[n];
    atomicAdd(pooled + (size_t)g * DIM + threadIdx.x, h[(size_t)n * DIM + threadIdx.x]);
    if (threadIdx.x == 0) atomicAdd(counts + g, 1.0f);
}

__global__ void pool_div_kernel(float* pooled, const float* counts, int G) {
    int i = blockIdx.x * blockDim.x + threadIdx.x;
    if (i >= G * DIM) return;
    pooled[i] /= fmaxf(counts[i >> 7], 1.0f);
}

// ---------------------------------------------------------------------------
// Host-side launch
// ---------------------------------------------------------------------------
static inline int ceil_div_i(long long a, long long b) { return (int)((a + b - 1) / b); }

// Convert A (f32 [M,K]) + W (f32 [K,N]) to f16 and run the WMMA GEMM.
static void gemm_f16(const float* A, const float* W, const float* bias, float* C,
                     _Float16* Ah, _Float16* Wt,
                     int M, int N, int K, int relu, hipStream_t stream) {
    int Kpad = (K + 31) & ~31;
    conv_pad_kernel<<<ceil_div_i((long long)M * Kpad, 256), 256, 0, stream>>>(A, Ah, M, K, Kpad);
    conv_trans_kernel<<<ceil_div_i((long long)N * Kpad, 256), 256, 0, stream>>>(W, Wt, K, N, Kpad);
    int waves = (M >> 4) * (N >> 6);
    wmma_gemm16_kernel<<<ceil_div_i(waves, 8), 256, 0, stream>>>(Ah, Wt, bias, C, M, N, Kpad, relu);
}

extern "C" void kernel_launch(void* const* d_in, const int* in_sizes, int n_in,
                              void* d_out, int out_size, void* d_ws, size_t ws_size,
                              hipStream_t stream) {
    const float* x       = (const float*)d_in[0];    // [N,78]
    const int*   eidx    = (const int*)  d_in[1];    // [2,E]
    const int*   batch   = (const int*)  d_in[2];    // [N]
    const float* W1      = (const float*)d_in[3];    // [78,128]
    const float* b1      = (const float*)d_in[4];
    const float* W2      = (const float*)d_in[5];    // [128,128]
    const float* b2      = (const float*)d_in[6];
    const float* lin_w   = (const float*)d_in[7];    // [L,128,4,128] -> per-layer [128,512]
    const float* att_src = (const float*)d_in[8];    // [L,4,128]
    const float* att_dst = (const float*)d_in[9];
    const float* conv_b  = (const float*)d_in[10];   // [L,128]
    const float* gW1     = (const float*)d_in[11];   // [128,1024]
    const float* gb1     = (const float*)d_in[12];
    const float* gW2     = (const float*)d_in[13];   // [1024,128]
    const float* gb2     = (const float*)d_in[14];

    const int Nn = in_sizes[0] / NODE_DIM;           // 50000
    const int E  = in_sizes[1] / 2;                  // 800000
    const int G  = out_size / DIM;                   // 1024
    const int* src = eidx;
    const int* dst = eidx + E;

    // Workspace layout (float units)
    float* ws     = (float*)d_ws;
    float* h      = ws;                                // N*128
    float* hp     = h      + (size_t)Nn * DIM;         // N*512 (also node-MLP temp)
    float* outac  = hp     + (size_t)Nn * HEADS * DIM; // N*512
    float* a_s    = outac  + (size_t)Nn * HEADS * DIM; // N*4
    float* a_d    = a_s    + (size_t)Nn * HEADS;
    int*   m_bits = (int*)(a_d + (size_t)Nn * HEADS);  // N*4
    float* denom  = (float*)(m_bits + (size_t)Nn * HEADS);
    float* pooled = denom  + (size_t)Nn * HEADS;       // G*128
    float* counts = pooled + (size_t)G * DIM;          // G
    float* hidden = counts + G;                        // G*1024
    // f16 scratch: Ah sized for max(M*Kpad) = N*128 halfs; Wt for max N*Kpad = 128K halfs
    _Float16* Ah  = (_Float16*)(hidden + (size_t)G * HID);
    _Float16* Wt  = Ah + (size_t)Nn * DIM;
    (void)ws_size;

    // ---- node MLP: h = relu(x@W1+b1)@W2+b2  (hp used as f32 temp) ----
    gemm_f16(x,  W1, b1, hp, Ah, Wt, Nn, DIM, NODE_DIM, 1, stream);
    gemm_f16(hp, W2, b2, h,  Ah, Wt, Nn, DIM, DIM,      0, stream);

    const long long eThreads = (long long)(E + Nn) * HEADS;
    for (int l = 0; l < LAYERS; ++l) {
        // projection: hp[N,512] = h @ lin_w[l]
        gemm_f16(h, lin_w + (size_t)l * DIM * HEADS * DIM, nullptr, hp,
                 Ah, Wt, Nn, HEADS * DIM, DIM, 0, stream);
        // attention coefficients
        att_coef_kernel<<<ceil_div_i((long long)Nn * HEADS * 32, 256), 256, 0, stream>>>(
            hp, att_src + (size_t)l * HEADS * DIM, att_dst + (size_t)l * HEADS * DIM,
            a_s, a_d, Nn);
        // reset segment state
        fill_i32_kernel<<<ceil_div_i((long long)Nn * HEADS, 256), 256, 0, stream>>>(
            m_bits, 0xFF800000, (size_t)Nn * HEADS);               // -inf
        fill_f32_kernel<<<ceil_div_i((long long)Nn * HEADS, 256), 256, 0, stream>>>(
            denom, 0.0f, (size_t)Nn * HEADS);
        fill_f32_kernel<<<ceil_div_i((long long)Nn * HEADS * DIM, 256), 256, 0, stream>>>(
            outac, 0.0f, (size_t)Nn * HEADS * DIM);
        // softmax + aggregation over edges (incl. self loops)
        edge_max_kernel<<<ceil_div_i(eThreads, 256), 256, 0, stream>>>(
            src, dst, E, Nn, a_s, a_d, m_bits);
        edge_denom_kernel<<<ceil_div_i(eThreads, 256), 256, 0, stream>>>(
            src, dst, E, Nn, a_s, a_d, m_bits, denom);
        edge_acc_kernel<<<E + Nn, 128, 0, stream>>>(
            src, dst, E, Nn, a_s, a_d, m_bits, denom, hp, outac);
        // head mean + bias -> h
        head_mean_kernel<<<ceil_div_i((long long)Nn * DIM, 256), 256, 0, stream>>>(
            outac, conv_b + (size_t)l * DIM, h, Nn);
    }

    // ---- global mean pool ----
    fill_f32_kernel<<<ceil_div_i((long long)G * DIM, 256), 256, 0, stream>>>(
        pooled, 0.0f, (size_t)G * DIM);
    fill_f32_kernel<<<ceil_div_i(G, 256), 256, 0, stream>>>(counts, 0.0f, (size_t)G);
    pool_acc_kernel<<<Nn, 128, 0, stream>>>(h, batch, pooled, counts, Nn);
    pool_div_kernel<<<ceil_div_i((long long)G * DIM, 256), 256, 0, stream>>>(pooled, counts, G);

    // ---- global FC head: out = relu(pooled@gW1+gb1)@gW2+gb2 ----
    gemm_f16(pooled, gW1, gb1, hidden, Ah, Wt, G, HID, DIM, 1, stream);
    gemm_f16(hidden, gW2, gb2, (float*)d_out, Ah, Wt, G, DIM, HID, 0, stream);
}
